// NeuralRecurrentDynamicsModel_80461917324013
// MI455X (gfx1250) — matrix operand order repaired and checked by reading, not compiled, not run on hardware
//
#include <hip/hip_runtime.h>
#include <hip/hip_bf16.h>
#include <math.h>

// ---------------- static problem dims ----------------
#define BSZ   2048
#define OBS   2048
#define HID   4096
#define ACTD  2048
#define PREM  8192      // OBS+HID+ACTD
#define PPM   4096
#define SEQ   8
#define DM    512
#define DI    1024
#define DS    16
#define DTR   32
#define KCV   4
#define BL    (BSZ*SEQ) // 16384 rows for mamba GEMMs

typedef __attribute__((ext_vector_type(16))) __bf16 bf16x16;
typedef __attribute__((ext_vector_type(8)))  float  f32x8;

// float -> bf16 (round to nearest even-ish), packed pair
__device__ __forceinline__ uint32_t pack_bf16x2(float a, float b) {
    union { float f; uint32_t u; } ua, ub;
    ua.f = a; ub.f = b;
    uint32_t lo = (ua.u + 0x7FFFu + ((ua.u >> 16) & 1u)) >> 16;
    uint32_t hi = (ub.u + 0x7FFFu + ((ub.u >> 16) & 1u)) >> 16;
    return (lo & 0xFFFFu) | (hi << 16);
}

// -----------------------------------------------------------------------------
// Staging: each thread t fetches 16 fp32 of one row (row = t>>1, k-half = t&1)
// and packs them into two uint4 of bf16 (c0 = k..k+7, c1 = k+8..k+15).
// Works for any block size: blockDim/2 rows per stage.
// -----------------------------------------------------------------------------
__device__ __forceinline__ void fetch_pack(const float* __restrict__ src, int ld,
                                           int row0, int k0, int tid,
                                           uint4& c0, uint4& c1) {
    int row = tid >> 1;
    int kk  = (tid & 1) << 4;
    const float* p = src + (size_t)(row0 + row) * ld + (k0 + kk);
    float4 f0 = ((const float4*)p)[0];
    float4 f1 = ((const float4*)p)[1];
    float4 f2 = ((const float4*)p)[2];
    float4 f3 = ((const float4*)p)[3];
    c0.x = pack_bf16x2(f0.x, f0.y); c0.y = pack_bf16x2(f0.z, f0.w);
    c0.z = pack_bf16x2(f1.x, f1.y); c0.w = pack_bf16x2(f1.z, f1.w);
    c1.x = pack_bf16x2(f2.x, f2.y); c1.y = pack_bf16x2(f2.z, f2.w);
    c1.z = pack_bf16x2(f3.x, f3.y); c1.w = pack_bf16x2(f3.z, f3.w);
}

// A-fragment layout (16x32 bf16, ISA 7.12.2): lane L holds row (L&15),
//   halves 0..7 -> K = 8*(L>>4)+0..7 ; halves 8..15 -> K = 16+8*(L>>4)+0..7
// Stored per 16x32 subtile as 32 lanes x 32 contiguous bytes (1 KiB/subtile).
__device__ __forceinline__ void commitA(uint32_t* smem, uint4 c0, uint4 c1, int tid) {
    int row = tid >> 1;
    int kk  = (tid & 1) << 4;
    int sub  = row >> 4;
    int lnLo = row & 15;
    uint4* dst = (uint4*)smem;
    dst[sub * 64 + lnLo * 2        + (kk >> 4)] = c0;   // k = kk..kk+7   -> lane lo
    dst[sub * 64 + (lnLo + 16) * 2 + (kk >> 4)] = c1;   // k = kk+8..+15  -> lane hi
}

// B-fragment layout (32x16 KxN bf16): lane L holds column (L&15), same K order as A:
// K = 8*(L>>4)+0..7, 16+8*(L>>4)+0..7; stored as plain row-major [16][32] bf16 per subtile.
__device__ __forceinline__ void commitB(uint32_t* smem, uint4 c0, uint4 c1, int tid) {
    int row = tid >> 1;
    int kk  = (tid & 1) << 4;
    uint4* dst = (uint4*)smem;
    int base = (row >> 4) * 64 + (row & 15) * 4 + (kk >> 3);
    dst[base]     = c0;
    dst[base + 1] = c1;
}

__device__ __forceinline__ bf16x16 fragA(const uint32_t* smem, int sub, int lane) {
    const uint4* p = (const uint4*)smem + sub * 64 + lane * 2;
    union { uint4 q[2]; bf16x16 v; } u;
    u.q[0] = p[0]; u.q[1] = p[1];
    return u.v;
}
__device__ __forceinline__ bf16x16 fragB(const uint32_t* smem, int sub, int lane) {
    const uint4* p = (const uint4*)smem + sub * 64 + (lane & 15) * 4 + (lane >> 4);
    union { uint4 q[2]; bf16x16 v; } u;
    u.q[0] = p[0]; u.q[1] = p[2];
    return u.v;
}

__device__ __forceinline__ float apply_act(float r, int ACT) {
    if (ACT == 1) r = fmaxf(r, 0.0f);
    else if (ACT == 2) r = (r > 20.0f) ? r : log1pf(__expf(r));
    return r;
}

// -----------------------------------------------------------------------------
// Large-tile GEMM: C[M,N] = act(A[M,K](lda) * W[N,K]^T + bias[N])
// 256 threads = 8 waves; tile 128x128x32, double-buffered LDS (2 x 16 KiB),
// global->reg prefetch pipelined over the WMMAs, ONE barrier per K-step.
// Each wave: 32x64 patch = 2 A-frags x 4 B-frags -> 8 v_wmma per K-step.
// Requires M%128==0, N%128==0, K%32==0.
// -----------------------------------------------------------------------------
template <int ACT>
__global__ void __launch_bounds__(256)
gemm128(const float* __restrict__ A, int lda,
        const float* __restrict__ W,
        const float* __restrict__ bias,
        float* __restrict__ C, int M, int N, int K) {
    __shared__ uint32_t As[2][2048];   // 8 KiB per stage (8 subtiles)
    __shared__ uint32_t Bs[2][2048];
    const int tid  = threadIdx.x;
    const int lane = tid & 31;
    const int wave = tid >> 5;
    const int bm = blockIdx.y * 128;
    const int bn = blockIdx.x * 128;
    const int wm = (wave >> 1) * 32;   // 0,32,64,96
    const int wn = (wave & 1) * 64;    // 0,64

    f32x8 acc[2][4] = {};

    uint4 ra0, ra1, rb0, rb1;
    fetch_pack(A, lda, bm, 0, tid, ra0, ra1);
    fetch_pack(W, K,  bn, 0, tid, rb0, rb1);
    commitA(As[0], ra0, ra1, tid);
    commitB(Bs[0], rb0, rb1, tid);
    __syncthreads();

    int p = 0;
    for (int k0 = 0; k0 < K; k0 += 32) {
        const bool more = (k0 + 32) < K;
        if (more) {                                 // prefetch next K-slice
            fetch_pack(A, lda, bm, k0 + 32, tid, ra0, ra1);
            fetch_pack(W, K,  bn, k0 + 32, tid, rb0, rb1);
        }
        bf16x16 afr[2], bfr[4];
        afr[0] = fragA(As[p], (wm >> 4) + 0, lane);
        afr[1] = fragA(As[p], (wm >> 4) + 1, lane);
#pragma unroll
        for (int j = 0; j < 4; ++j) bfr[j] = fragB(Bs[p], (wn >> 4) + j, lane);
#pragma unroll
        for (int i = 0; i < 2; ++i)
#pragma unroll
            for (int j = 0; j < 4; ++j)
                acc[i][j] = __builtin_amdgcn_wmma_f32_16x16x32_bf16(
                    false, afr[i], false, bfr[j], (short)0, acc[i][j], false, false);
        if (more) {                                 // commit next stage
            commitA(As[p ^ 1], ra0, ra1, tid);
            commitB(Bs[p ^ 1], rb0, rb1, tid);
            __syncthreads();
        }
        p ^= 1;
    }

    const int lhi  = lane >> 4;
    const int lcol = lane & 15;
#pragma unroll
    for (int i = 0; i < 2; ++i)
#pragma unroll
        for (int j = 0; j < 4; ++j) {
            int n = bn + wn + j * 16 + lcol;
            float bv = bias ? bias[n] : 0.0f;
            int mbase = bm + wm + i * 16 + lhi * 8;
#pragma unroll
            for (int v = 0; v < 8; ++v)
                C[(size_t)(mbase + v) * N + n] = apply_act(acc[i][j][v] + bv, ACT);
        }
}

// -----------------------------------------------------------------------------
// Small-tile GEMM fallback (x_proj N=64, any M%64/N%64/K%32 shape).
// 128 threads = 4 waves; tile 64x64x32; each wave 2x2 WMMA tiles.
// -----------------------------------------------------------------------------
template <int ACT>
__global__ void __launch_bounds__(128)
gemm64(const float* __restrict__ A, int lda,
       const float* __restrict__ W,
       const float* __restrict__ bias,
       float* __restrict__ C, int M, int N, int K) {
    __shared__ uint32_t As[1024];   // 4 KiB
    __shared__ uint32_t Bs[1024];
    const int tid  = threadIdx.x;
    const int lane = tid & 31;
    const int wave = tid >> 5;
    const int bm = blockIdx.y * 64;
    const int bn = blockIdx.x * 64;
    const int wm = (wave >> 1) * 32;
    const int wn = (wave & 1) * 32;

    f32x8 acc[2][2] = {};

    for (int k0 = 0; k0 < K; k0 += 32) {
        __syncthreads();
        uint4 ra0, ra1, rb0, rb1;
        fetch_pack(A, lda, bm, k0, tid, ra0, ra1);
        fetch_pack(W, K,  bn, k0, tid, rb0, rb1);
        commitA(As, ra0, ra1, tid);
        commitB(Bs, rb0, rb1, tid);
        __syncthreads();
        bf16x16 afr[2], bfr[2];
        afr[0] = fragA(As, (wm >> 4) + 0, lane);
        afr[1] = fragA(As, (wm >> 4) + 1, lane);
        bfr[0] = fragB(Bs, (wn >> 4) + 0, lane);
        bfr[1] = fragB(Bs, (wn >> 4) + 1, lane);
#pragma unroll
        for (int i = 0; i < 2; ++i)
#pragma unroll
            for (int j = 0; j < 2; ++j)
                acc[i][j] = __builtin_amdgcn_wmma_f32_16x16x32_bf16(
                    false, afr[i], false, bfr[j], (short)0, acc[i][j], false, false);
    }

    const int lhi  = lane >> 4;
    const int lcol = lane & 15;
#pragma unroll
    for (int i = 0; i < 2; ++i)
#pragma unroll
        for (int j = 0; j < 2; ++j) {
            int n = bn + wn + j * 16 + lcol;
            float bv = bias ? bias[n] : 0.0f;
            int mbase = bm + wm + i * 16 + lhi * 8;
#pragma unroll
            for (int v = 0; v < 8; ++v)
                C[(size_t)(mbase + v) * N + n] = apply_act(acc[i][j][v] + bv, ACT);
        }
}

// ---------------- concat([obs, h, action]) -> xcat [B, 8192] ----------------
__global__ void concat_kernel(const float* __restrict__ obs, const float* __restrict__ h,
                              const float* __restrict__ act, float* __restrict__ xcat) {
    int idx = blockIdx.x * blockDim.x + threadIdx.x;
    if (idx >= BSZ * PREM) return;
    int row = idx / PREM, col = idx - row * PREM;
    float v;
    if (col < OBS)             v = obs[(size_t)row * OBS + col];
    else if (col < OBS + HID)  v = h  [(size_t)row * HID + (col - OBS)];
    else                       v = act[(size_t)row * ACTD + (col - OBS - HID)];
    xcat[idx] = v;
}

// ---------------- GRU gates (biases already added in the two GEMMs) ----------
__global__ void gru_gate_kernel(const float* __restrict__ gi, const float* __restrict__ gh,
                                const float* __restrict__ h, float* __restrict__ hout) {
    int idx = blockIdx.x * blockDim.x + threadIdx.x;
    if (idx >= BSZ * HID) return;
    int b = idx / HID, c = idx - b * HID;
    size_t ro = (size_t)b * (3 * HID);
    float ir = gi[ro + c], iz = gi[ro + HID + c], in_ = gi[ro + 2 * HID + c];
    float hr = gh[ro + c], hz = gh[ro + HID + c], hn  = gh[ro + 2 * HID + c];
    float r = 1.0f / (1.0f + __expf(-(ir + hr)));
    float z = 1.0f / (1.0f + __expf(-(iz + hz)));
    float n = tanhf(in_ + r * hn);
    hout[idx] = (1.0f - z) * n + z * h[idx];
}

// ---------------- depthwise causal conv1d (K=4) + SiLU -----------------------
__global__ void conv_silu_kernel(const float* __restrict__ xi, const float* __restrict__ cw,
                                 const float* __restrict__ cb, float* __restrict__ xc) {
    int idx = blockIdx.x * blockDim.x + threadIdx.x;
    if (idx >= BSZ * SEQ * DI) return;
    int d = idx & (DI - 1);
    int s = (idx >> 10) & (SEQ - 1);
    int b = idx >> 13;
    const float* base = xi + (size_t)b * SEQ * DI + d;
    float acc = cb[d];
#pragma unroll
    for (int t = 0; t < KCV; ++t) {
        int ss = s - (KCV - 1) + t;
        if (ss >= 0) acc += cw[d * KCV + t] * base[(size_t)ss * DI];
    }
    xc[idx] = acc / (1.0f + __expf(-acc));   // x * sigmoid(x)
}

// ---------------- selective scan over L=8, fused D-skip + SiLU(z) gating -----
__global__ void mamba_scan_kernel(const float* __restrict__ xc, const float* __restrict__ dt,
                                  const float* __restrict__ xdbl, const float* __restrict__ z,
                                  const float* __restrict__ A_log, const float* __restrict__ Dp,
                                  float* __restrict__ y) {
    int idx = blockIdx.x * blockDim.x + threadIdx.x;
    if (idx >= BSZ * DI) return;
    int d = idx & (DI - 1);
    int b = idx >> 10;
    float aneg[DS], h[DS];
#pragma unroll
    for (int n = 0; n < DS; ++n) { aneg[n] = -__expf(A_log[d * DS + n]); h[n] = 0.0f; }
    float Dv = Dp[d];
    for (int s = 0; s < SEQ; ++s) {
        size_t o  = ((size_t)b * SEQ + s) * DI + d;
        size_t bo = ((size_t)b * SEQ + s) * (DTR + 2 * DS);
        float dtv = dt[o], xcv = xc[o];
        float yv = 0.0f;
#pragma unroll
        for (int n = 0; n < DS; ++n) {
            float dA  = __expf(dtv * aneg[n]);
            float Bn  = xdbl[bo + DTR + n];
            float Cn  = xdbl[bo + DTR + DS + n];
            h[n] = dA * h[n] + dtv * Bn * xcv;
            yv  += h[n] * Cn;
        }
        float zv = z[o];
        float sz = zv / (1.0f + __expf(-zv));
        y[o] = (yv + Dv * xcv) * sz;
    }
}

// ---------------- host side ------------------------------------------------
static inline void run_gemm(int act, const float* A, int lda, const float* W,
                            const float* bias, float* C, int M, int N, int K,
                            hipStream_t s) {
    if ((M % 128 == 0) && (N % 128 == 0)) {
        dim3 g(N / 128, M / 128), b(256);
        if (act == 0)      gemm128<0><<<g, b, 0, s>>>(A, lda, W, bias, C, M, N, K);
        else if (act == 1) gemm128<1><<<g, b, 0, s>>>(A, lda, W, bias, C, M, N, K);
        else               gemm128<2><<<g, b, 0, s>>>(A, lda, W, bias, C, M, N, K);
    } else {
        dim3 g(N / 64, M / 64), b(128);
        if (act == 0)      gemm64<0><<<g, b, 0, s>>>(A, lda, W, bias, C, M, N, K);
        else if (act == 1) gemm64<1><<<g, b, 0, s>>>(A, lda, W, bias, C, M, N, K);
        else               gemm64<2><<<g, b, 0, s>>>(A, lda, W, bias, C, M, N, K);
    }
}

extern "C" void kernel_launch(void* const* d_in, const int* in_sizes, int n_in,
                              void* d_out, int out_size, void* d_ws, size_t ws_size,
                              hipStream_t stream) {
    (void)in_sizes; (void)n_in; (void)out_size; (void)ws_size;
    // ---- inputs (setup_inputs insertion order) ----
    const float* obs = (const float*)d_in[0];
    const float* hst = (const float*)d_in[1];
    const float* act = (const float*)d_in[2];
    const float* mW[4] = {(const float*)d_in[3], (const float*)d_in[5],
                          (const float*)d_in[7], (const float*)d_in[9]};
    const float* mB[4] = {(const float*)d_in[4], (const float*)d_in[6],
                          (const float*)d_in[8], (const float*)d_in[10]};
    const float* w_ih = (const float*)d_in[11];
    const float* w_hh = (const float*)d_in[12];
    const float* b_ih = (const float*)d_in[13];
    const float* b_hh = (const float*)d_in[14];
    const float* mlp1W = (const float*)d_in[42];
    const float* mlp1b = (const float*)d_in[43];

    // ---- workspace overlay (peak ~416 MiB) ----
#define WSF(mib) ((float*)((char*)d_ws + (size_t)(mib) * (1u << 20)))
    float* XCAT = WSF(0);     // 64 MiB  (freed after mlp0 L0)
    float* ACTA = WSF(64);    // 32 MiB
    float* ACTB = WSF(96);    // 32 MiB  (final mlp0 out == mamba layer0 input)
    float* GI   = WSF(128);   // 96 MiB  (GRU scratch, reused by mamba below)
    float* GH   = WSF(224);   // 96 MiB
    // mamba scratch (valid after GRU gates consumed):
    float* XI   = WSF(0);     // 64 MiB  (reuses XCAT)
    float* ZB   = WSF(128);   // 64 MiB  (reuses GI)
    float* XDBL = WSF(192);   // 4  MiB  (reuses GI tail)
    float* XC   = WSF(224);   // 64 MiB  (reuses GH)
    float* DT   = WSF(288);   // 64 MiB
    float* YB   = WSF(0);     // 64 MiB  (reuses XI after conv consumed it)
    float* XM1  = WSF(352);   // 32 MiB
    float* XM2  = WSF(384);   // 32 MiB

    float* out_obs = (float*)d_out;                         // [2048,2048]
    float* out_h   = (float*)d_out + (size_t)BSZ * OBS;     // [2048,4096]

    const int TB = 256;

    // ---- concat + mlp0 (relu on first 3 layers) ----
    concat_kernel<<<(BSZ * PREM + TB - 1) / TB, TB, 0, stream>>>(obs, hst, act, XCAT);
    run_gemm(1, XCAT, PREM, mW[0], mB[0], ACTA, BSZ, PPM, PREM, stream);
    run_gemm(1, ACTA, PPM,  mW[1], mB[1], ACTB, BSZ, PPM, PPM,  stream);
    run_gemm(1, ACTB, PPM,  mW[2], mB[2], ACTA, BSZ, PPM, PPM,  stream);
    run_gemm(0, ACTA, PPM,  mW[3], mB[3], ACTB, BSZ, PPM, PPM,  stream);

    // ---- GRUCell: h_hat ----
    run_gemm(0, ACTB, PPM, w_ih, b_ih, GI, BSZ, 3 * HID, PPM, stream);
    run_gemm(0, hst,  HID, w_hh, b_hh, GH, BSZ, 3 * HID, HID, stream);
    gru_gate_kernel<<<(BSZ * HID + TB - 1) / TB, TB, 0, stream>>>(GI, GH, hst, out_h);

    // ---- 3 stacked Mamba blocks over [B, 8, 512] ----
    const float* xin = ACTB;             // reshape is a no-op (row-major)
    float* xouts[3] = {XM1, XM2, XM1};
    for (int l = 0; l < 3; ++l) {
        const int base = 15 + 9 * l;
        const float* in_proj = (const float*)d_in[base + 0];   // [2*DI, DM]
        const float* conv_w  = (const float*)d_in[base + 1];   // [DI, 4]
        const float* conv_b  = (const float*)d_in[base + 2];   // [DI]
        const float* x_proj  = (const float*)d_in[base + 3];   // [64, DI]
        const float* dt_w    = (const float*)d_in[base + 4];   // [DI, 32]
        const float* dt_b    = (const float*)d_in[base + 5];   // [DI]
        const float* A_log   = (const float*)d_in[base + 6];   // [DI, 16]
        const float* Dm      = (const float*)d_in[base + 7];   // [DI]
        const float* outp    = (const float*)d_in[base + 8];   // [DM, DI]
        float* xout = xouts[l];

        // in_proj split into xi / z halves (rows [0,DI) and [DI,2DI) of W)
        run_gemm(0, xin, DM, in_proj,                   nullptr, XI, BL, DI, DM, stream);
        run_gemm(0, xin, DM, in_proj + (size_t)DI * DM, nullptr, ZB, BL, DI, DM, stream);
        // depthwise causal conv + SiLU
        conv_silu_kernel<<<(BSZ * SEQ * DI + TB - 1) / TB, TB, 0, stream>>>(XI, conv_w, conv_b, XC);
        // x_proj -> [dt(32) | B(16) | C(16)]  (N=64 -> small-tile kernel)
        run_gemm(0, XC, DI, x_proj, nullptr, XDBL, BL, DTR + 2 * DS, DI, stream);
        // dt = softplus(xdbl[:, :32] @ dt_w^T + dt_b)   (lda = 64, K = 32)
        run_gemm(2, XDBL, DTR + 2 * DS, dt_w, dt_b, DT, BL, DI, DTR, stream);
        // selective scan + D-skip + SiLU(z) gating
        mamba_scan_kernel<<<(BSZ * DI + TB - 1) / TB, TB, 0, stream>>>(XC, DT, XDBL, ZB, A_log, Dm, YB);
        // out_proj
        run_gemm(0, YB, DI, outp, nullptr, xout, BL, DM, DI, stream);
        xin = xout;
    }

    // ---- mlp1: obs_latent_hat ----
    run_gemm(0, XM1, PPM, mlp1W, mlp1b, out_obs, BSZ, OBS, PPM, stream);
}
